// MRTE_17154099380769
// MI455X (gfx1250) — matrix operand order repaired and checked by reading, not compiled, not run on hardware
//
#include <hip/hip_runtime.h>
#include <hip/hip_bf16.h>
#include <math.h>

// ---------------------------------------------------------------------------
// MI455X / gfx1250, wave32. All matrix math via v_wmma_f32_16x16x32_bf16.
// All WMMA operands are pre-packed into bf16 buffers whose layouts match the
// ISA 7.12.2 fragment layouts exactly, so the steady-state inner loops are
// pure {global_load_b128 x8 -> v_wmma x4} with no converts.
// ---------------------------------------------------------------------------

typedef __attribute__((ext_vector_type(16))) __bf16 bf16x16;
typedef __attribute__((ext_vector_type(8)))  __bf16 bf16x8;
typedef __attribute__((ext_vector_type(8)))  float  floatx8;

namespace {
constexpr int cB    = 8;
constexpr int cLP   = 256;
constexpr int cTM   = 1024;
constexpr int cTG   = 256;
constexpr int cMEL  = 80;
constexpr int cMELP = 96;          // 80 padded to multiple of 32
constexpr int cH    = 512;
constexpr int cNH   = 8;
constexpr int cDH   = cH / cNH;    // 64
constexpr int cLC   = cLP + cTG;   // 512
constexpr int cTOUT = cLC * 3;     // 1536
constexpr int cKG   = 31;
}

__device__ __forceinline__ floatx8 wmma_bf16(bf16x16 a, bf16x16 b, floatx8 c) {
  // (neg_a, A, neg_b, B, c_mod, C, reuse_a, reuse_b)
  return __builtin_amdgcn_wmma_f32_16x16x32_bf16(false, a, false, b, (short)0, c,
                                                 false, false);
}

// ===========================================================================
// One-time packing kernels (bandwidth-trivial).
// ===========================================================================

// f32 -> bf16, same layout.
__global__ void __launch_bounds__(256) pack_bf16_kernel(
    const float* __restrict__ src, __bf16* __restrict__ dst, int n) {
  const int i = (int)(blockIdx.x * blockDim.x + threadIdx.x);
  if (i < n) dst[i] = (__bf16)src[i];
}

// f32 channel-major [nB][C][T] -> bf16 time-major zero-padded [nB*T][Cp].
__global__ void __launch_bounds__(256) transpose_pad_kernel(
    const float* __restrict__ src, __bf16* __restrict__ dst,
    int nB, int C, int T, int Cp) {
  const int i = (int)(blockIdx.x * blockDim.x + threadIdx.x);
  if (i >= nB * T * Cp) return;
  const int c = i % Cp, bt = i / Cp, t = bt % T, b = bt / T;
  dst[i] = (c < C) ? (__bf16)src[((size_t)b * C + c) * T + t] : (__bf16)0.f;
}

// f32 [Cout][Cin][Kw] -> bf16 [Kw][Cout][Cp] (ci contiguous, zero-padded).
__global__ void __launch_bounds__(256) pack_convw_kernel(
    const float* __restrict__ w, __bf16* __restrict__ wp,
    int Cout, int Cin, int Kw, int Cp) {
  const int i = (int)(blockIdx.x * blockDim.x + threadIdx.x);
  if (i >= Kw * Cout * Cp) return;
  const int c = i % Cp, r = i / Cp, co = r % Cout, dk = r / Cout;
  wp[i] = (c < Cin) ? (__bf16)w[((size_t)co * Cin + c) * Kw + dk] : (__bf16)0.f;
}

// ===========================================================================
// Implicit-im2col conv1d ('same' padding), all-bf16 operands.
//   in_tm : bf16 [nB*T][Cp]      (time-major activations, zero-padded ci)
//   wp    : bf16 [Kw][Cout][Cp]  (fragment-native weights)
//   out   : bf16 time-major [nB*T][Cout]  (out_f32=0)  -- chains into next conv
//           f32  time-major [nB*T][Cout]  (out_f32=1)  -- final layer
// One wave = 32(co) x 32(b,t); steady state = 8x b128 loads + 4x wmma.
// ===========================================================================
__global__ void __launch_bounds__(256) conv_wmma_kernel(
    const __bf16* __restrict__ in_tm, const __bf16* __restrict__ wp,
    const float* __restrict__ bias, void* __restrict__ outp,
    int nB, int Cp, int Cout, int T, int Kw, int relu, int out_f32)
{
  const int wave = (int)((blockIdx.x * blockDim.x + threadIdx.x) >> 5);
  const int lane = (int)(threadIdx.x & 31);
  const int nWN = (nB * T) >> 5;
  const int wm = wave / nWN, wn = wave % nWN;
  if (wm * 32 >= Cout) return;
  const int co0 = wm * 32, n0 = wn * 32;
  const int hl = lane >> 4, lr = lane & 15;
  const int pad = Kw >> 1;

  floatx8 acc[2][2] = {};
  for (int dk = 0; dk < Kw; ++dk) {
    if (dk + 1 < Kw)  // cover L2 latency of the next dk weight slab
      __builtin_prefetch(wp + ((size_t)(dk + 1) * Cout + co0 + lr) * Cp, 0, 3);
    for (int ci0 = 0; ci0 < Cp; ci0 += 32) {
      bf16x16 af[2], bfr[2];
#pragma unroll
      for (int t = 0; t < 2; ++t) {
        // A frag: co = co0+t*16+lr, k(ci) = i + 8*((i>>3)+half) -> 2 runs of 8
        const __bf16* ap = wp + ((size_t)dk * Cout + co0 + t * 16 + lr) * Cp + ci0;
        const bf16x8 alo = *(const bf16x8*)(ap + 8 * hl);
        const bf16x8 ahi = *(const bf16x8*)(ap + 16 + 8 * hl);
#pragma unroll
        for (int i = 0; i < 8; ++i) { af[t][i] = alo[i]; af[t][i + 8] = ahi[i]; }
        // B frag: col (b,t) = n0+t*16+lr, k(ci) = i + 8*((i>>3)+half) -> 2 runs of 8
        const int n = n0 + t * 16 + lr;
        const int bb = n / T, tt = n % T;
        const int ts = tt + dk - pad;
        bf16x8 blo = {}, bhi = {};
        if (ts >= 0 && ts < T) {
          const __bf16* ip = in_tm + ((size_t)bb * T + ts) * Cp + ci0 + 8 * hl;
          blo = *(const bf16x8*)ip;
          bhi = *(const bf16x8*)(ip + 16);
        }
#pragma unroll
        for (int i = 0; i < 8; ++i) { bfr[t][i] = blo[i]; bfr[t][i + 8] = bhi[i]; }
      }
#pragma unroll
      for (int a = 0; a < 2; ++a)
#pragma unroll
        for (int b2 = 0; b2 < 2; ++b2)
          acc[a][b2] = wmma_bf16(af[a], bfr[b2], acc[a][b2]);
    }
  }
  // epilogue: 8 consecutive co per lane -> one 16B store (bf16) / merged f32
#pragma unroll
  for (int a = 0; a < 2; ++a)
#pragma unroll
    for (int b2 = 0; b2 < 2; ++b2) {
      const int n = n0 + b2 * 16 + lr;
      const int bb = n / T, tt = n % T;
      const int cobase = co0 + a * 16 + 8 * hl;
      if (out_f32) {
        float* o = (float*)outp + ((size_t)bb * T + tt) * Cout + cobase;
#pragma unroll
        for (int j = 0; j < 8; ++j) {
          float vv = acc[a][b2][j] + bias[cobase + j];
          o[j] = relu ? fmaxf(vv, 0.f) : vv;
        }
      } else {
        bf16x8 pk;
#pragma unroll
        for (int j = 0; j < 8; ++j) {
          float vv = acc[a][b2][j] + bias[cobase + j];
          pk[j] = (__bf16)(relu ? fmaxf(vv, 0.f) : vv);
        }
        *(bf16x8*)((__bf16*)outp + ((size_t)bb * T + tt) * Cout + cobase) = pk;
      }
    }
}

// ===========================================================================
// GEMM: C = A[M,K] * W[N,K]^T + bias, bf16 operands.
// mode 0: bf16 out [M][N]   (q, k, ctx inputs for later WMMA stages)
// mode 1: bf16 out [N][M]   (transposed -- V for attention PV fragments)
// mode 2: f32  out [M][N]   (attn_out for the gather)
// ===========================================================================
__global__ void __launch_bounds__(256) gemm_wmma_kernel(
    const __bf16* __restrict__ A, const __bf16* __restrict__ W,
    const float* __restrict__ bias, void* __restrict__ outp,
    int M, int N, int K, int mode)
{
  const int wave = (int)((blockIdx.x * blockDim.x + threadIdx.x) >> 5);
  const int lane = (int)(threadIdx.x & 31);
  const int nWN = N >> 5;
  const int wm = wave / nWN, wn = wave % nWN;
  if (wm * 32 >= M) return;
  const int row0 = wm * 32, col0 = wn * 32;
  const int hl = lane >> 4, lr = lane & 15;

  floatx8 acc[2][2] = {};
  for (int k0 = 0; k0 < K; k0 += 32) {
    bf16x16 af[2], bfr[2];
#pragma unroll
    for (int t = 0; t < 2; ++t) {
      const __bf16* ap = A + (size_t)(row0 + t * 16 + lr) * K + k0;
      const bf16x8 alo = *(const bf16x8*)(ap + 8 * hl);
      const bf16x8 ahi = *(const bf16x8*)(ap + 16 + 8 * hl);
#pragma unroll
      for (int i = 0; i < 8; ++i) { af[t][i] = alo[i]; af[t][i + 8] = ahi[i]; }
      const __bf16* wptr = W + (size_t)(col0 + t * 16 + lr) * K + k0 + 8 * hl;
      const bf16x8 blo = *(const bf16x8*)wptr;
      const bf16x8 bhi = *(const bf16x8*)(wptr + 16);
#pragma unroll
      for (int i = 0; i < 8; ++i) { bfr[t][i] = blo[i]; bfr[t][i + 8] = bhi[i]; }
    }
#pragma unroll
    for (int a = 0; a < 2; ++a)
#pragma unroll
      for (int b2 = 0; b2 < 2; ++b2)
        acc[a][b2] = wmma_bf16(af[a], bfr[b2], acc[a][b2]);
  }
#pragma unroll
  for (int a = 0; a < 2; ++a)
#pragma unroll
    for (int b2 = 0; b2 < 2; ++b2) {
      const int c = col0 + b2 * 16 + lr;
      const float bv = bias ? bias[c] : 0.f;
      if (mode == 1) {          // bf16 [N][M]: 8 consecutive rows -> 16B store
        bf16x8 pk;
#pragma unroll
        for (int j = 0; j < 8; ++j) pk[j] = (__bf16)(acc[a][b2][j] + bv);
        *(bf16x8*)((__bf16*)outp + (size_t)c * M + row0 + a * 16 + 8 * hl) = pk;
      } else {
#pragma unroll
        for (int j = 0; j < 8; ++j) {
          const int r = row0 + a * 16 + j + 8 * hl;
          const float vv = acc[a][b2][j] + bv;
          if (mode == 2) ((float*)outp)[(size_t)r * N + c] = vv;
          else           ((__bf16*)outp)[(size_t)r * N + c] = (__bf16)vv;
        }
      }
    }
}

// ===========================================================================
// Flash attention, one wave per (b, head, 16-q tile), DH=64, all-bf16 inputs.
// Computes S^T = K x Q^T so the softmaxed P tile repacks into the PV WMMA's
// A-fragment with zero cross-lane movement (per-lane key sets line up).
// q,k: bf16 [B*rows][H] row-major; vt: bf16 [H][B*TM] transposed.
// ===========================================================================
__global__ void __launch_bounds__(256) attention_kernel(
    const __bf16* __restrict__ q, const __bf16* __restrict__ k,
    const __bf16* __restrict__ vt, __bf16* __restrict__ ctx)
{
  const int wave = (int)((blockIdx.x * blockDim.x + threadIdx.x) >> 5);
  const int lane = (int)(threadIdx.x & 31);
  const int qtiles = cLP / 16;
  const int b  = wave / (qtiles * cNH);
  const int h  = (wave / qtiles) % cNH;
  const int q0 = (wave % qtiles) * 16;
  if (b >= cB) return;
  const int hl = lane >> 4, lr = lane & 15;
  const float scale = 0.125f;                      // 1/sqrt(DH)

  bf16x16 qf[2];                                   // Q^T B-fragments over d
#pragma unroll
  for (int t = 0; t < 2; ++t) {
    const __bf16* qp =
        q + ((size_t)b * cLP + q0 + lr) * cH + h * cDH + t * 32 + 8 * hl;
    const bf16x8 lo = *(const bf16x8*)qp;
    const bf16x8 hi = *(const bf16x8*)(qp + 16);
#pragma unroll
    for (int i = 0; i < 8; ++i) { qf[t][i] = lo[i]; qf[t][i + 8] = hi[i]; }
  }

  floatx8 acc[4] = {};
  float m_i = -INFINITY, l_i = 0.f;

  for (int key0 = 0; key0 < cTM; key0 += 32) {
    floatx8 st[2];
#pragma unroll
    for (int c = 0; c < 2; ++c) {
      floatx8 s = {};
#pragma unroll
      for (int t = 0; t < 2; ++t) {
        bf16x16 kf;                                // A frag from K rows
        const __bf16* kp =
            k + ((size_t)b * cTM + key0 + c * 16 + lr) * cH + h * cDH + t * 32;
        const bf16x8 alo = *(const bf16x8*)(kp + 8 * hl);
        const bf16x8 ahi = *(const bf16x8*)(kp + 16 + 8 * hl);
#pragma unroll
        for (int i = 0; i < 8; ++i) { kf[i] = alo[i]; kf[i + 8] = ahi[i]; }
        s = wmma_bf16(kf, qf[t], s);
      }
      st[c] = s;
    }
    // online softmax (scaled scores); stats per q-row live at q = lane&15
    float mloc = -INFINITY;
#pragma unroll
    for (int c = 0; c < 2; ++c)
#pragma unroll
      for (int j = 0; j < 8; ++j) mloc = fmaxf(mloc, st[c][j] * scale);
    mloc = fmaxf(mloc, __shfl_xor(mloc, 16, 32));
    const float mnew = fmaxf(m_i, mloc);
    const float corr = __expf(m_i - mnew);
    float psum = 0.f;
#pragma unroll
    for (int c = 0; c < 2; ++c)
#pragma unroll
      for (int j = 0; j < 8; ++j) {
        const float p = __expf(st[c][j] * scale - mnew);
        st[c][j] = p;
        psum += p;
      }
    psum += __shfl_xor(psum, 16, 32);
    l_i = l_i * corr + psum;
    m_i = mnew;

    float corrs[8];                                // ctx rows live at j+8*half
#pragma unroll
    for (int j = 0; j < 8; ++j) corrs[j] = __shfl(corr, j + 8 * hl, 32);
#pragma unroll
    for (int t = 0; t < 4; ++t)
#pragma unroll
      for (int j = 0; j < 8; ++j) acc[t][j] *= corrs[j];

    bf16x16 pf;                                    // P A-fragment, in-lane repack
#pragma unroll
    for (int i = 0; i < 16; ++i)
      pf[i] = (__bf16)((i < 8) ? st[0][i] : st[1][i - 8]);

#pragma unroll
    for (int t = 0; t < 4; ++t) {
      bf16x16 vf;                                  // B frag: keys 8h+i, 16+8h+i
      const __bf16* vp = vt + (size_t)(h * cDH + t * 16 + lr) * (cB * cTM) +
                         (size_t)b * cTM + key0 + 8 * hl;
      const bf16x8 blo = *(const bf16x8*)vp;
      const bf16x8 bhi = *(const bf16x8*)(vp + 16);
#pragma unroll
      for (int i = 0; i < 8; ++i) { vf[i] = blo[i]; vf[i + 8] = bhi[i]; }
      acc[t] = wmma_bf16(pf, vf, acc[t]);
    }
  }

  float linv[8];
#pragma unroll
  for (int j = 0; j < 8; ++j) linv[j] = 1.f / __shfl(l_i, j + 8 * hl, 32);
#pragma unroll
  for (int t = 0; t < 4; ++t)
#pragma unroll
    for (int j = 0; j < 8; ++j)
      ctx[((size_t)b * cLP + q0 + j + 8 * hl) * cH + h * cDH + t * 16 + lr] =
          (__bf16)(acc[t][j] * linv[j]);
}

// ===========================================================================
// Length regulator: cumsum -> t->l map -> row gather. Pure bandwidth.
// ===========================================================================
__global__ void __launch_bounds__(256) regulate_kernel(
    const int* __restrict__ tl, const float* __restrict__ attn_out,
    const float* __restrict__ g_tm, float* __restrict__ out)
{
  const int b = blockIdx.x;
  const int tid = (int)threadIdx.x;
  __shared__ int   s_len[cLC];
  __shared__ int   s_cum[cLC];
  __shared__ short s_map[cTOUT];
  for (int l = tid; l < cLC; l += blockDim.x) s_len[l] = tl[b * cLC + l];
  for (int t = tid; t < cTOUT; t += blockDim.x) s_map[t] = -1;
  __syncthreads();
  if (tid == 0) {
    int s = 0;
    for (int l = 0; l < cLC; ++l) { s += s_len[l]; s_cum[l] = s; }
  }
  __syncthreads();
  for (int l = tid; l < cLC; l += blockDim.x) {
    const int e = s_cum[l];
    for (int t = e - s_len[l]; t < e; ++t) s_map[t] = (short)l;
  }
  __syncthreads();
  float4* op = (float4*)(out + (size_t)b * cTOUT * cH);
  const int HQ = cH / 4;
  for (int idx = tid; idx < cTOUT * HQ; idx += blockDim.x) {
    const int t = idx / HQ, hq = idx % HQ;
    const int l = s_map[t];
    float4 val = make_float4(0.f, 0.f, 0.f, 0.f);
    if (l >= 0) {
      const float* src = (l < cLP)
                             ? (attn_out + ((size_t)b * cLP + l) * cH)
                             : (g_tm + ((size_t)b * cTG + (l - cLP)) * cH);
      val = ((const float4*)src)[hq];
    }
    op[idx] = val;
  }
}

// ===========================================================================
// Host orchestration
// ===========================================================================
extern "C" void kernel_launch(void* const* d_in, const int* in_sizes, int n_in,
                              void* d_out, int out_size, void* d_ws, size_t ws_size,
                              hipStream_t stream) {
  (void)in_sizes; (void)n_in; (void)out_size; (void)ws_size;
  const float* phone      = (const float*)d_in[0];   // [B,LP,H]
  const float* mel_spec   = (const float*)d_in[1];   // [B,MEL,TM]
  const float* gmel       = (const float*)d_in[2];   // [B,MEL,TG]
  const int*   tlen       = (const int*)  d_in[3];   // [B,LC]
  const float* mel_conv_w = (const float*)d_in[4];   // [H,MEL,3]
  const float* mel_conv_b = (const float*)d_in[5];
  const float* in_proj_w  = (const float*)d_in[6];   // [3H,H]
  const float* in_proj_b  = (const float*)d_in[7];
  const float* out_proj_w = (const float*)d_in[8];   // [H,H]
  const float* out_proj_b = (const float*)d_in[9];
  const float* g0_w       = (const float*)d_in[10];  // [H,MEL,31]
  const float* g0_b       = (const float*)d_in[11];
  const float* g_ws_      = (const float*)d_in[12];  // [4,H,H,31]
  const float* g_bs       = (const float*)d_in[13];  // [4,H]
  float* out = (float*)d_out;

  char* ws = (char*)d_ws;
  size_t off = 0;
  auto carveB = [&](size_t elems) {                  // bf16
    char* p = ws + off; off += (elems * 2 + 255) & ~size_t(255); return (__bf16*)p;
  };
  auto carveF = [&](size_t elems) {                  // f32
    char* p = ws + off; off += (elems * 4 + 255) & ~size_t(255); return (float*)p;
  };

  __bf16* wqkv   = carveB((size_t)3 * cH * cH);          // packed in_proj
  __bf16* wo     = carveB((size_t)cH * cH);              // packed out_proj
  __bf16* phoneb = carveB((size_t)cB * cLP * cH);
  __bf16* mels_t = carveB((size_t)cB * cTM * cMELP);     // mel_spec time-major
  __bf16* gmel_t = carveB((size_t)cB * cTG * cMELP);
  __bf16* melw_p = carveB((size_t)3 * cH * cMELP);       // [3][H][96]
  __bf16* g0w_p  = carveB((size_t)cKG * cH * cMELP);     // [31][H][96]
  __bf16* gw_p   = carveB((size_t)4 * cKG * cH * cH);    // [4][31][H][H] (65MB)
  __bf16* melb   = carveB((size_t)cB * cTM * cH);        // mel, time-major bf16
  __bf16* qb     = carveB((size_t)cB * cLP * cH);
  __bf16* kb     = carveB((size_t)cB * cTM * cH);
  __bf16* vtb    = carveB((size_t)cB * cTM * cH);        // transposed [H][B*TM]
  __bf16* ctxb   = carveB((size_t)cB * cLP * cH);
  float*  attno  = carveF((size_t)cB * cLP * cH);
  __bf16* gp0    = carveB((size_t)cB * cTG * cH);
  __bf16* gp1    = carveB((size_t)cB * cTG * cH);
  float*  gtm    = carveF((size_t)cB * cTG * cH);

  const dim3 blk(256);
  auto ew = [](size_t n) { return dim3((unsigned)((n + 255) / 256)); };
  auto tile_grid = [](int Mt, int Nt) {
    return dim3((unsigned)(((Mt / 32) * (Nt / 32) + 7) / 8));
  };

  // ---- one-time packs ----
  pack_bf16_kernel<<<ew((size_t)3 * cH * cH), blk, 0, stream>>>(in_proj_w, wqkv, 3 * cH * cH);
  pack_bf16_kernel<<<ew((size_t)cH * cH), blk, 0, stream>>>(out_proj_w, wo, cH * cH);
  pack_bf16_kernel<<<ew((size_t)cB * cLP * cH), blk, 0, stream>>>(phone, phoneb, cB * cLP * cH);
  transpose_pad_kernel<<<ew((size_t)cB * cTM * cMELP), blk, 0, stream>>>(
      mel_spec, mels_t, cB, cMEL, cTM, cMELP);
  transpose_pad_kernel<<<ew((size_t)cB * cTG * cMELP), blk, 0, stream>>>(
      gmel, gmel_t, cB, cMEL, cTG, cMELP);
  pack_convw_kernel<<<ew((size_t)3 * cH * cMELP), blk, 0, stream>>>(
      mel_conv_w, melw_p, cH, cMEL, 3, cMELP);
  pack_convw_kernel<<<ew((size_t)cKG * cH * cMELP), blk, 0, stream>>>(
      g0_w, g0w_p, cH, cMEL, cKG, cMELP);
  for (int i = 0; i < 4; ++i)
    pack_convw_kernel<<<ew((size_t)cKG * cH * cH), blk, 0, stream>>>(
        g_ws_ + (size_t)i * cH * cH * cKG, gw_p + (size_t)i * cKG * cH * cH,
        cH, cH, cKG, cH);

  // ---- 1) mel = conv1d(mel_spec, k=3) -> bf16 time-major [B*TM][H] ----
  conv_wmma_kernel<<<tile_grid(cH, cB * cTM), blk, 0, stream>>>(
      mels_t, melw_p, mel_conv_b, melb, cB, cMELP, cH, cTM, 3, 0, 0);

  // ---- 2) Q/K/V projections (V transposed for PV fragments) ----
  gemm_wmma_kernel<<<tile_grid(cB * cLP, cH), blk, 0, stream>>>(
      phoneb, wqkv, in_proj_b, qb, cB * cLP, cH, cH, 0);
  gemm_wmma_kernel<<<tile_grid(cB * cTM, cH), blk, 0, stream>>>(
      melb, wqkv + (size_t)cH * cH, in_proj_b + cH, kb, cB * cTM, cH, cH, 0);
  gemm_wmma_kernel<<<tile_grid(cB * cTM, cH), blk, 0, stream>>>(
      melb, wqkv + (size_t)2 * cH * cH, in_proj_b + 2 * cH, vtb,
      cB * cTM, cH, cH, 1);

  // ---- 3) flash attention -> ctx bf16 ----
  attention_kernel<<<dim3((unsigned)((cB * cNH * (cLP / 16) + 7) / 8)), blk, 0,
                     stream>>>(qb, kb, vtb, ctxb);

  // ---- 4) output projection -> f32 attn_out ----
  gemm_wmma_kernel<<<tile_grid(cB * cLP, cH), blk, 0, stream>>>(
      ctxb, wo, out_proj_b, attno, cB * cLP, cH, cH, 2);

  // ---- 5) g stack: g0 (80->512) + 4x (512->512), k=31, ReLU ----
  const dim3 ggrid = tile_grid(cH, cB * cTG);
  conv_wmma_kernel<<<ggrid, blk, 0, stream>>>(
      gmel_t, g0w_p, g0_b, gp0, cB, cMELP, cH, cTG, cKG, 1, 0);
  const __bf16* src = gp0;
  __bf16* pong[2] = { gp1, gp0 };
  for (int i = 0; i < 4; ++i) {
    void* dst = (i == 3) ? (void*)gtm : (void*)pong[i & 1];
    conv_wmma_kernel<<<ggrid, blk, 0, stream>>>(
        src, gw_p + (size_t)i * cKG * cH * cH, g_bs + (size_t)i * cH, dst,
        cB, cH, cH, cTG, cKG, 1, (i == 3) ? 1 : 0);
    if (i < 3) src = pong[i & 1];
  }

  // ---- 6) length-regulator gather -> out [B, T_OUT, H] ----
  regulate_kernel<<<dim3(cB), blk, 0, stream>>>(tlen, attno, gtm, out);
}
